// CrossAttention_51797305589990
// MI455X (gfx1250) — matrix operand order repaired and checked by reading, not compiled, not run on hardware
//
#include <hip/hip_runtime.h>

#define HEADS 16
#define DHEAD 64
#define INNER 1024          // HEADS * DHEAD
#define BATCH 4
#define NQ    2048          // query sequence length
#define MC    1024          // context sequence length
#define QDIM  1024
#define CDIM  768

typedef __attribute__((ext_vector_type(16))) __bf16        v16bf;
typedef __attribute__((ext_vector_type(8)))  float         v8f;
typedef __attribute__((ext_vector_type(4)))  unsigned int  v4u;

union Frag16 { v4u q[2]; v16bf v; };
union Pack2  { __bf16 h[2]; unsigned int u; };
union Pack8  { __bf16 h[8]; v4u u; };

__device__ __forceinline__ v8f wmma_bf16(v16bf a, v16bf b, v8f c) {
  return __builtin_amdgcn_wmma_f32_16x16x32_bf16(false, a, false, b, (short)0, c,
                                                 false, false);
}

// ---------------------------------------------------------------------------
// Elementwise f32 -> bf16, 8 elements per thread (b128 in/out)
// ---------------------------------------------------------------------------
__global__ __launch_bounds__(256) void cvt_bf16_kernel(const float* __restrict__ in,
                                                       __bf16* __restrict__ out,
                                                       int n8) {
  int i = blockIdx.x * 256 + threadIdx.x;
  if (i < n8) {
    const float* p = in + (size_t)i * 8;
    Pack8 o;
#pragma unroll
    for (int e = 0; e < 8; ++e) o.h[e] = (__bf16)p[e];
    *(v4u*)(out + (size_t)i * 8) = o.u;
  }
}

// W [K x C] row-major f32  ->  Wt [C x K] row-major bf16
__global__ __launch_bounds__(256) void cvt_tr_kernel(const float* __restrict__ W,
                                                     __bf16* __restrict__ Wt,
                                                     int K, int C) {
  int i = blockIdx.x * 256 + threadIdx.x;
  if (i < K * C) {
    int k = i / C, c = i % C;
    Wt[(size_t)c * K + k] = (__bf16)W[i];
  }
}

// ---------------------------------------------------------------------------
// WMMA GEMM: C[rows x cols] = A[rows x K](bf16, row-major) * Bt^T
//   Bt is [cols x K] bf16 (B column-major) -> contiguous 32B B-fragments.
//   8 waves / block, 64x64 tile per wave (4x4 WMMA).
//   K is a template constant: the double-buffered K loop fully unrolls and all
//   fragment loads use immediate instruction offsets from 8 base pointers.
// MODE 0: store bf16 row-major (ldc)
// MODE 1: store bf16 scattered as Vt[b][h][d][m]
// MODE 2: store f32 row-major with bias
// ---------------------------------------------------------------------------
template <int MODE, int K>
__global__ __launch_bounds__(256) void gemm_wmma_kernel(
    const __bf16* __restrict__ A, const __bf16* __restrict__ Bt,
    void* __restrict__ C, const float* __restrict__ bias, int ldc) {
  const int lane = threadIdx.x & 31;
  const int wave = threadIdx.x >> 5;
  const int l16  = lane & 15;
  const int h8   = lane >> 4;
  const int row0 = blockIdx.x * 256 + (wave & 3) * 64;
  const int col0 = blockIdx.y * 128 + (wave >> 2) * 64;

  const __bf16* ap[4];
  const __bf16* bp[4];
#pragma unroll
  for (int i = 0; i < 4; ++i)
    ap[i] = A + (size_t)(row0 + i * 16 + l16) * K + h8 * 16;
#pragma unroll
  for (int j = 0; j < 4; ++j)
    bp[j] = Bt + (size_t)(col0 + j * 16 + l16) * K + h8 * 16;

  v8f acc[4][4] = {};
  Frag16 aA[4], aB[4];
  v16bf  bA[4], bB[4];

#pragma unroll
  for (int i = 0; i < 4; ++i) {
    aA[i].q[0] = *(const v4u*)(ap[i]);
    aA[i].q[1] = *(const v4u*)(ap[i] + 8);
  }
#pragma unroll
  for (int j = 0; j < 4; ++j) bA[j] = *(const v16bf*)(bp[j]);

#pragma unroll
  for (int k0 = 0; k0 < K; k0 += 64) {
    // prefetch k0+32 (always in-range: K % 64 == 0)
#pragma unroll
    for (int i = 0; i < 4; ++i) {
      aB[i].q[0] = *(const v4u*)(ap[i] + k0 + 32);
      aB[i].q[1] = *(const v4u*)(ap[i] + k0 + 40);
    }
#pragma unroll
    for (int j = 0; j < 4; ++j) bB[j] = *(const v16bf*)(bp[j] + k0 + 32);

#pragma unroll
    for (int i = 0; i < 4; ++i)
#pragma unroll
      for (int j = 0; j < 4; ++j)
        acc[i][j] = wmma_bf16(aA[i].v, bA[j], acc[i][j]);

    if (k0 + 64 < K) {   // compile-time under full unroll
#pragma unroll
      for (int i = 0; i < 4; ++i) {
        aA[i].q[0] = *(const v4u*)(ap[i] + k0 + 64);
        aA[i].q[1] = *(const v4u*)(ap[i] + k0 + 72);
      }
#pragma unroll
      for (int j = 0; j < 4; ++j) bA[j] = *(const v16bf*)(bp[j] + k0 + 64);
    }

#pragma unroll
    for (int i = 0; i < 4; ++i)
#pragma unroll
      for (int j = 0; j < 4; ++j)
        acc[i][j] = wmma_bf16(aB[i].v, bB[j], acc[i][j]);
  }

#pragma unroll
  for (int i = 0; i < 4; ++i) {
#pragma unroll
    for (int j = 0; j < 4; ++j) {
#pragma unroll
      for (int r = 0; r < 8; ++r) {
        const int row = row0 + i * 16 + h8 * 8 + r;
        const int col = col0 + j * 16 + l16;
        if (MODE == 0) {
          ((__bf16*)C)[(size_t)row * ldc + col] = (__bf16)acc[i][j][r];
        } else if (MODE == 1) {
          const int bb = row >> 10, m = row & 1023;     // MC == 1024
          const int hh = col >> 6,  d = col & 63;
          ((__bf16*)C)[(((size_t)(bb * HEADS + hh) * DHEAD + d) << 10) + m] =
              (__bf16)acc[i][j][r];
        } else {
          ((float*)C)[(size_t)row * ldc + col] = acc[i][j][r] + bias[col];
        }
      }
    }
  }
}

// ---------------------------------------------------------------------------
// Attention: one block per (b, h, 64 query rows). 8 waves.
//   S[64 x 1024] f32 in LDS (~258KB of the 320KB WGP LDS).
//   Pass 1: S = Q*K^T            (WMMA, fully unrolled, immediate offsets)
//   Pass 2: rowwise exp(x-max),  packed bf16 written in place (deferred norm)
//   Pass 3: O = P*V (WMMA, A-frags = 2x ds_load_b128, fully unrolled),
//           O scaled by 1/rowsum at the epilogue.
// ---------------------------------------------------------------------------
#define SP 1032  // padded S row stride (floats), keeps 16B alignment

__global__ __launch_bounds__(256) void attn_kernel(
    const __bf16* __restrict__ q_bf, const __bf16* __restrict__ k_bf,
    const __bf16* __restrict__ vt_bf, __bf16* __restrict__ o_bf) {
  extern __shared__ float smem[];
  float* S    = smem;               // 64 * SP
  float* red  = smem + 64 * SP;     // 256 (max partials, then 64 row sums)
  float* red2 = red + 256;          // 256 (sum partials)

  const int nt = blockIdx.x & 31;
  const int h  = (blockIdx.x >> 5) & 15;
  const int b  = blockIdx.x >> 9;
  const int n0 = nt * 64;

  const int lane = threadIdx.x & 31;
  const int wave = threadIdx.x >> 5;
  const int l16  = lane & 15;
  const int h8   = lane >> 4;

  // ----- Pass 1: S = Q K^T -------------------------------------------------
  {
    const int rt    = wave & 3;              // 16-row tile within 64 rows
    const int cbase = (wave >> 2) * 512;     // column half of M

    Frag16 aq[2];
    const __bf16* qrow =
        q_bf + (size_t)(b * NQ + n0 + rt * 16 + l16) * INNER + h * DHEAD;
#pragma unroll
    for (int kc = 0; kc < 2; ++kc) {
      const __bf16* pa = qrow + kc * 32 + h8 * 16;
      aq[kc].q[0] = *(const v4u*)pa;
      aq[kc].q[1] = *(const v4u*)(pa + 8);
    }

    // one base pointer; tile displacement ct*16*INNER folds into load offsets
    const __bf16* kbase = k_bf + (size_t)(b * MC + cbase + l16) * INNER +
                          h * DHEAD + h8 * 16;
    float* srow[8];
#pragma unroll
    for (int r = 0; r < 8; ++r)
      srow[r] = S + (size_t)(rt * 16 + h8 * 8 + r) * SP + cbase + l16;

    v16bf c0k0 = *(const v16bf*)(kbase);
    v16bf c0k1 = *(const v16bf*)(kbase + 32);

#pragma unroll
    for (int ct = 0; ct < 32; ct += 2) {
      const int o1 = (ct + 1) * 16 * INNER;
      v16bf c1k0 = *(const v16bf*)(kbase + o1);
      v16bf c1k1 = *(const v16bf*)(kbase + o1 + 32);

      v8f acc = {};
      acc = wmma_bf16(aq[0].v, c0k0, acc);
      acc = wmma_bf16(aq[1].v, c0k1, acc);
#pragma unroll
      for (int r = 0; r < 8; ++r) srow[r][ct * 16] = acc[r];

      if (ct + 2 < 32) {   // compile-time under full unroll
        const int o2 = (ct + 2) * 16 * INNER;
        c0k0 = *(const v16bf*)(kbase + o2);
        c0k1 = *(const v16bf*)(kbase + o2 + 32);
      }

      v8f acc2 = {};
      acc2 = wmma_bf16(aq[0].v, c1k0, acc2);
      acc2 = wmma_bf16(aq[1].v, c1k1, acc2);
#pragma unroll
      for (int r = 0; r < 8; ++r) srow[r][ct * 16 + 16] = acc2[r];
    }
  }
  __syncthreads();

  // ----- Pass 2: exp(scale*(x-max)), pack bf16 in place, defer 1/sum -------
  {
    const int row = threadIdx.x >> 2;   // 64 rows, 4 threads each
    const int seg = threadIdx.x & 3;
    float* sr = S + (size_t)row * SP;
    const float* f = sr + seg * 256;

    float m = -3.4e38f;
    for (int c = 0; c < 256; ++c) m = fmaxf(m, f[c]);
    red[threadIdx.x] = m;
    __syncthreads();
    const float rm = fmaxf(fmaxf(red[row * 4 + 0], red[row * 4 + 1]),
                           fmaxf(red[row * 4 + 2], red[row * 4 + 3]));

    // pack bf16 pairs into the first half of this thread's own dword range:
    // write dword (seg*256 + t/2) always trails read dword (seg*256 + t).
    unsigned int* pu = (unsigned int*)sr;
    float sum = 0.f;
    for (int t = 0; t < 256; t += 2) {
      float p0 = __expf((f[t]     - rm) * 0.125f);
      float p1 = __expf((f[t + 1] - rm) * 0.125f);
      sum += p0 + p1;
      Pack2 pk;
      pk.h[0] = (__bf16)p0;
      pk.h[1] = (__bf16)p1;
      pu[seg * 256 + (t >> 1)] = pk.u;
    }
    red2[threadIdx.x] = sum;
    __syncthreads();
    if (threadIdx.x < 64)
      red[threadIdx.x] = red2[threadIdx.x * 4 + 0] + red2[threadIdx.x * 4 + 1] +
                         red2[threadIdx.x * 4 + 2] + red2[threadIdx.x * 4 + 3];
  }
  __syncthreads();

  // ----- Pass 3: O = P V, scaled by 1/rowsum at store ----------------------
  {
    const int rt = wave & 3;       // 16-row tile
    const int cp = wave >> 2;      // which 32-wide d half
    v8f acc0 = {}, acc1 = {};
    // packed P element index for key k: 512*(k>>8) + (k&255)  (bf16 units)
    const __bf16* prow_bf = (const __bf16*)(S + (size_t)(rt * 16 + l16) * SP);
    const __bf16* v0 = vt_bf +
        ((size_t)(b * HEADS + h) * DHEAD + cp * 32 + l16) * MC + h8 * 16;
    const __bf16* v1 = v0 + (size_t)16 * MC;

    v16bf cb0 = *(const v16bf*)(v0);
    v16bf cb1 = *(const v16bf*)(v1);

#pragma unroll
    for (int k0 = 0; k0 < MC; k0 += 64) {
      v16bf nb0 = *(const v16bf*)(v0 + k0 + 32);
      v16bf nb1 = *(const v16bf*)(v1 + k0 + 32);

      Frag16 ap;
      const int e0 = 512 * (k0 >> 8) + (k0 & 255) + h8 * 16;
      ap.q[0] = *(const v4u*)(prow_bf + e0);
      ap.q[1] = *(const v4u*)(prow_bf + e0 + 8);
      acc0 = wmma_bf16(ap.v, cb0, acc0);
      acc1 = wmma_bf16(ap.v, cb1, acc1);

      if (k0 + 64 < MC) {   // compile-time under full unroll
        cb0 = *(const v16bf*)(v0 + k0 + 64);
        cb1 = *(const v16bf*)(v1 + k0 + 64);
      }

      const int k1 = k0 + 32;
      const int e1 = 512 * (k1 >> 8) + (k1 & 255) + h8 * 16;
      ap.q[0] = *(const v4u*)(prow_bf + e1);
      ap.q[1] = *(const v4u*)(prow_bf + e1 + 8);
      acc0 = wmma_bf16(ap.v, nb0, acc0);
      acc1 = wmma_bf16(ap.v, nb1, acc1);
    }

#pragma unroll
    for (int r = 0; r < 8; ++r) {
      const int rl  = rt * 16 + h8 * 8 + r;
      const float inv = 1.0f / red[rl];
      __bf16* po = o_bf + (size_t)(b * NQ + n0 + rl) * INNER + h * DHEAD;
      po[cp * 32 + l16]      = (__bf16)(acc0[r] * inv);
      po[cp * 32 + 16 + l16] = (__bf16)(acc1[r] * inv);
    }
  }
}

// ---------------------------------------------------------------------------
extern "C" void kernel_launch(void* const* d_in, const int* in_sizes, int n_in,
                              void* d_out, int out_size, void* d_ws, size_t ws_size,
                              hipStream_t stream) {
  (void)in_sizes; (void)n_in; (void)out_size; (void)ws_size;
  const float* x   = (const float*)d_in[0];
  const float* ctx = (const float*)d_in[1];
  const float* Wq  = (const float*)d_in[2];
  const float* Wk  = (const float*)d_in[3];
  const float* Wv  = (const float*)d_in[4];
  const float* Wo  = (const float*)d_in[5];
  const float* bo  = (const float*)d_in[6];

  char* ws = (char*)d_ws;
  size_t off = 0;
  auto alloc = [&](size_t bytes) -> void* {
    void* p = ws + off;
    off += (bytes + 255) & ~(size_t)255;
    return p;
  };

  const int n_x   = BATCH * NQ * QDIM;   // 8,388,608
  const int n_ctx = BATCH * MC * CDIM;   // 3,145,728

  __bf16* x_bf  = (__bf16*)alloc((size_t)n_x * 2);
  __bf16* c_bf  = (__bf16*)alloc((size_t)n_ctx * 2);
  __bf16* Wqt   = (__bf16*)alloc((size_t)QDIM * INNER * 2);
  __bf16* Wkt   = (__bf16*)alloc((size_t)CDIM * INNER * 2);
  __bf16* Wvt   = (__bf16*)alloc((size_t)CDIM * INNER * 2);
  __bf16* Wot   = (__bf16*)alloc((size_t)INNER * QDIM * 2);
  __bf16* q_bf  = (__bf16*)alloc((size_t)BATCH * NQ * INNER * 2);
  __bf16* k_bf  = (__bf16*)alloc((size_t)BATCH * MC * INNER * 2);
  __bf16* vt_bf = (__bf16*)alloc((size_t)BATCH * MC * INNER * 2);
  __bf16* o_bf  = (__bf16*)alloc((size_t)BATCH * NQ * INNER * 2);

  // --- conversions ---
  cvt_bf16_kernel<<<(n_x / 8 + 255) / 256, 256, 0, stream>>>(x, x_bf, n_x / 8);
  cvt_bf16_kernel<<<(n_ctx / 8 + 255) / 256, 256, 0, stream>>>(ctx, c_bf, n_ctx / 8);
  cvt_tr_kernel<<<(QDIM * INNER + 255) / 256, 256, 0, stream>>>(Wq, Wqt, QDIM, INNER);
  cvt_tr_kernel<<<(CDIM * INNER + 255) / 256, 256, 0, stream>>>(Wk, Wkt, CDIM, INNER);
  cvt_tr_kernel<<<(CDIM * INNER + 255) / 256, 256, 0, stream>>>(Wv, Wvt, CDIM, INNER);
  cvt_tr_kernel<<<(INNER * QDIM + 255) / 256, 256, 0, stream>>>(Wo, Wot, INNER, QDIM);

  // --- projections ---
  {
    dim3 g(BATCH * NQ / 256, INNER / 128);       // 32 x 8
    gemm_wmma_kernel<0, QDIM><<<g, 256, 0, stream>>>(x_bf, Wqt, (void*)q_bf,
                                                     nullptr, INNER);
  }
  {
    dim3 g(BATCH * MC / 256, INNER / 128);       // 16 x 8
    gemm_wmma_kernel<0, CDIM><<<g, 256, 0, stream>>>(c_bf, Wkt, (void*)k_bf,
                                                     nullptr, INNER);
    gemm_wmma_kernel<1, CDIM><<<g, 256, 0, stream>>>(c_bf, Wvt, (void*)vt_bf,
                                                     nullptr, INNER);
  }

  // --- attention ---
  {
    const size_t shmem = (size_t)(64 * SP + 512) * sizeof(float);  // ~266 KB
    dim3 g(BATCH * HEADS * (NQ / 64));           // 2048 blocks
    attn_kernel<<<g, 256, shmem, stream>>>(q_bf, k_bf, vt_bf, o_bf);
  }

  // --- output projection + bias (f32 out) ---
  {
    dim3 g(BATCH * NQ / 256, QDIM / 128);        // 32 x 8
    gemm_wmma_kernel<2, INNER><<<g, 256, 0, stream>>>(o_bf, Wot, d_out, bo, QDIM);
  }
}
